// ST_block_6803228197043
// MI455X (gfx1250) — matrix operand order repaired and checked by reading, not compiled, not run on hardware
//
#include <hip/hip_runtime.h>
#include <hip/hip_bf16.h>

// ---------- problem constants ----------
#define B_ 4
#define N_ 1024
#define TT 12
#define TPAD 16
#define HH 8
#define DK_ 32
#define CC 256
#define SITES 2          // (b,n) sites per workgroup -> M = 24 rows
#define NTHREADS 256     // 8 wave32 waves

typedef __attribute__((ext_vector_type(16))) __bf16 v16bf;
typedef __attribute__((ext_vector_type(8)))  float  v8f;

union BFrag {
  v16bf v;
  unsigned u[8];
  __bf16 h[16];
};

// ---------- WMMA wrapper: D = A(16x32 bf16) * B(32x16 bf16) + C(f32) ----------
__device__ __forceinline__ v8f wmma_bf16(v16bf a, v16bf b, v8f c) {
  return __builtin_amdgcn_wmma_f32_16x16x32_bf16(
      /*neg_a=*/false, a, /*neg_b=*/false, b,
      /*c_mod=*/(short)0, c, /*reuse_a=*/false, /*reuse_b=*/false);
}

// A-fragment (16x32, bf16) from row-major [m][k] storage (LDS or global).
// ISA layout: lanes 0-15 hold K[k0+0..7]+K[k0+16..23]; lanes 16-31 hold +8.
__device__ __forceinline__ v16bf load_a_bf16(const __bf16* base, int ldk,
                                             int m, int k0, int lane) {
  int lh = (lane >> 4) & 1;
  const __bf16* row = base + m * ldk;
  const unsigned* p0 = (const unsigned*)(row + k0 + 8 * lh);
  const unsigned* p1 = (const unsigned*)(row + k0 + 16 + 8 * lh);
  BFrag f;
  f.u[0] = p0[0]; f.u[1] = p0[1]; f.u[2] = p0[2]; f.u[3] = p0[3];
  f.u[4] = p1[0]; f.u[5] = p1[1]; f.u[6] = p1[2]; f.u[7] = p1[3];
  return f.v;
}

// B-fragment (32x16, bf16) from row-major [n][k] storage.
// ISA layout: lane n=l%16; lanes 0-15 hold K[k0+0..7]+K[k0+16..23]; lanes 16-31 hold +8.
// Per lane this is two runs of 8 bf16 = two b128 loads.
__device__ __forceinline__ v16bf load_b_bf16(const __bf16* base, int ldk,
                                             int n0, int k0, int lane) {
  int n = n0 + (lane & 15);
  int lh = (lane >> 4) & 1;
  const unsigned* p = (const unsigned*)(base + n * ldk + k0 + 8 * lh);
  BFrag f;
#pragma unroll
  for (int i = 0; i < 8; ++i) f.u[i] = p[(i & 3) + 8 * (i >> 2)];
  return f.v;
}

// A-fragment from fp32 global rows with on-the-fly bf16 convert; zero pad rows.
__device__ __forceinline__ v16bf load_a_f32cvt(const float* base, int ldk,
                                               int m, int k0, int lane,
                                               bool valid) {
  int lh = (lane >> 4) & 1;
  BFrag f;
  if (valid) {
    const float* row = base + (long)m * ldk;
#pragma unroll
    for (int i = 0; i < 8; ++i) f.h[i]     = (__bf16)row[k0 + 8 * lh + i];
#pragma unroll
    for (int i = 0; i < 8; ++i) f.h[8 + i] = (__bf16)row[k0 + 16 + 8 * lh + i];
  } else {
#pragma unroll
    for (int i = 0; i < 16; ++i) f.h[i] = (__bf16)0.f;
  }
  return f.v;
}

// ---------- LDS layout helpers ----------
// PROJ: [7 slots][SITES][HH][TPAD][DK]  (slots: Qf,Kf,Vf,Qs,Ks,Vs,QFS)
// VT:   [2 fs][SITES][HH][DK][32]      (V transposed, s padded to 32 w/ zeros)
// CTX:  [4 att][SITES][TPAD][CC]
// PS:   [8 waves][TPAD][32]            (softmaxed P, per-wave scratch)
#define PROJ_ELEMS (7 * SITES * HH * TPAD * DK_)
#define VT_ELEMS   (2 * SITES * HH * DK_ * 32)
#define CTX_ELEMS  (4 * SITES * TPAD * CC)
#define PS_ELEMS   (8 * TPAD * 32)
#define SMEM_ELEMS (PROJ_ELEMS + VT_ELEMS + CTX_ELEMS + PS_ELEMS)
#define SMEM_BYTES (SMEM_ELEMS * 2)

__device__ __forceinline__ int proj_at(int slot, int site, int h, int t, int d) {
  return (((slot * SITES + site) * HH + h) * TPAD + t) * DK_ + d;
}
__device__ __forceinline__ int vt_at(int fs, int site, int h, int d, int s) {
  return (((fs * SITES + site) * HH + h) * DK_ + d) * 32 + s;
}
__device__ __forceinline__ int ctx_at(int att, int site, int t, int c) {
  return ((att * SITES + site) * TPAD + t) * CC + c;
}

// ---------- weight fp32 -> bf16 pre-pass ----------
struct WPtrs { const float* p[7]; };

__global__ void convert_weights_kernel(WPtrs w, __bf16* out) {
  long i = (long)blockIdx.x * blockDim.x + threadIdx.x;
  if (i < 7L * CC * CC) {
    int wi  = (int)(i >> 16);
    int off = (int)(i & 65535);
    out[i] = (__bf16)w.p[wi][off];
  }
}

// ---------- fused FlowSpeed kernel ----------
__global__ __launch_bounds__(NTHREADS)
void flowspeed_fused_kernel(const float* __restrict__ x0, const float* __restrict__ x1,
                            const float* __restrict__ x2, const float* __restrict__ x3,
                            const float* __restrict__ x4, const float* __restrict__ x5,
                            const __bf16* __restrict__ wts,   // 7 x [256][256] bf16
                            const float* __restrict__ b_out,
                            const float* __restrict__ kj,
                            const float* __restrict__ vf_fs,
                            float* __restrict__ out) {
  extern __shared__ char smem_raw[];
  __bf16* PROJ = (__bf16*)smem_raw;
  __bf16* VT   = PROJ + PROJ_ELEMS;
  __bf16* CTX  = VT + VT_ELEMS;
  __bf16* PS   = CTX + CTX_ELEMS;

  const int tid  = threadIdx.x;
  const int lane = tid & 31;
  const int wave = tid >> 5;           // 0..7
  const int wg   = blockIdx.x;         // handles sites [wg*2, wg*2+1]
  const int lh   = (lane >> 4) & 1;
  const int col  = lane & 15;

  // Zero all LDS: guarantees K-dim padding (t=12..15 rows, s=12..31 cols) is 0.
  {
    unsigned* sp = (unsigned*)smem_raw;
#pragma unroll 4
    for (int i = tid; i < SMEM_ELEMS / 2; i += NTHREADS) sp[i] = 0u;
  }
  __syncthreads();

  // ---------------- Phase 1: six input projections  Y = X * W^T -------------
  // 24 jobs = tensor(6) x mtile(2) x nhalf(2); each wave takes 3.
  const float* xin[6] = {x0, x1, x2, x3, x4, x5};
  const long rowbase = (long)wg * SITES * TT;  // 24 consecutive token rows

  for (int job = wave; job < 24; job += 8) {
    int tensor = job >> 2;
    int mtile  = (job >> 1) & 1;
    int nhalf  = job & 1;
    const float*  X = xin[tensor] + rowbase * CC;
    const __bf16* W = wts + tensor * CC * CC;

    v8f acc[8];
#pragma unroll
    for (int i = 0; i < 8; ++i) acc[i] = v8f{0.f, 0.f, 0.f, 0.f, 0.f, 0.f, 0.f, 0.f};

    int m = mtile * 16 + col;
    bool mvalid = m < SITES * TT;

    for (int ks = 0; ks < 8; ++ks) {
      int k0 = ks * 32;
      v16bf a = load_a_f32cvt(X, CC, m, k0, lane, mvalid);
      if (ks < 7 && mvalid) __builtin_prefetch(X + (long)m * CC + k0 + 32, 0, 1);
#pragma unroll
      for (int nt = 0; nt < 8; ++nt) {
        int n0 = (nhalf * 8 + nt) * 16;
        v16bf b = load_b_bf16(W, CC, n0, k0, lane);
        acc[nt] = wmma_bf16(a, b, acc[nt]);
      }
    }
    // Scatter D tiles to LDS as bf16 per (site, head); V also transposed.
#pragma unroll
    for (int nt = 0; nt < 8; ++nt) {
      int c = (nhalf * 8 + nt) * 16 + col;
      int h = c >> 5, d = c & 31;
#pragma unroll
      for (int r = 0; r < 8; ++r) {
        int mm = mtile * 16 + r + 8 * lh;
        if (mm < SITES * TT) {
          int site = mm / TT, t = mm % TT;
          __bf16 bv = (__bf16)acc[nt][r];
          PROJ[proj_at(tensor, site, h, t, d)] = bv;
          if (tensor == 2) VT[vt_at(0, site, h, d, t)] = bv;  // Vf^T
          if (tensor == 5) VT[vt_at(1, site, h, d, t)] = bv;  // Vs^T
        }
      }
    }
  }
  __syncthreads();

  // ---------------- Phase 2: physics transform  QFS = kj*(Qs - Qs^2/(Vf+eps))
  for (int i = tid; i < SITES * HH * TT * DK_; i += NTHREADS) {
    int d    = i & 31;
    int t    = (i >> 5) % TT;
    int rem  = i / (32 * TT);
    int h    = rem & 7;
    int site = rem >> 3;
    float q  = (float)PROJ[proj_at(3, site, h, t, d)];
    float v  = kj[t] * (q - q * q / (vf_fs[t] + 1e-5f));
    PROJ[proj_at(6, site, h, t, d)] = (__bf16)v;
  }
  __syncthreads();

  // ---------------- Phase 3: four attentions per (site, head) ---------------
  // att0 z_ff = att(Qf,Kf)·Vf   att1 z_fs = att(Kf,QFS)·Vf
  // att2 z_sf = att(Ks,Qf)·Vs   att3 z_ss = att(Qs,Ks)·Vs
  const int aA[4] = {0, 1, 4, 3};
  const int aB[4] = {1, 6, 0, 4};
  const int aV[4] = {0, 0, 1, 1};
  const float scale = 0.17677669529663687f;  // 1/sqrt(DK)
  __bf16* myPS = PS + wave * TPAD * 32;

  for (int task = wave; task < 64; task += 8) {
    int att  = task & 3;
    int h    = (task >> 2) & 7;
    int site = task >> 5;

    // scores: S = Q(16x32) * K^T   (K stored [s][d] == B layout, K-dim = DK = 32)
    v16bf aq = load_a_bf16(PROJ + proj_at(aA[att], site, h, 0, 0), DK_, col, 0, lane);
    v16bf bk = load_b_bf16(PROJ + proj_at(aB[att], site, h, 0, 0), DK_, 0, 0, lane);
    v8f s = v8f{0.f, 0.f, 0.f, 0.f, 0.f, 0.f, 0.f, 0.f};
    s = wmma_bf16(aq, bk, s);

    // masked row softmax across the 16-lane half (D layout: col = lane%16)
    float p[8];
#pragma unroll
    for (int r = 0; r < 8; ++r) {
      float v = s[r] * scale;
      if (col >= TT) v = -3.0e38f;
      float mx = v;
#pragma unroll
      for (int off = 1; off < 16; off <<= 1)
        mx = fmaxf(mx, __shfl_xor(mx, off, 32));
      float e = (col < TT) ? __expf(v - mx) : 0.f;
      float sm = e;
#pragma unroll
      for (int off = 1; off < 16; off <<= 1)
        sm += __shfl_xor(sm, off, 32);
      p[r] = e / sm;
    }
    // stash P (bf16) into wave-private scratch [t][s]; s=12..31 stay zero
#pragma unroll
    for (int r = 0; r < 8; ++r) myPS[(r + 8 * lh) * 32 + col] = (__bf16)p[r];

    // ctx: O = P(16x32) * V(32x32)  via V^T[d][s]; two N-tiles of 16
    v16bf pa = load_a_bf16(myPS, 32, col, 0, lane);
    const __bf16* vtb = VT + vt_at(aV[att], site, h, 0, 0);
    v8f o0 = v8f{0.f, 0.f, 0.f, 0.f, 0.f, 0.f, 0.f, 0.f};
    v8f o1 = o0;
    o0 = wmma_bf16(pa, load_b_bf16(vtb, 32, 0, 0, lane), o0);
    o1 = wmma_bf16(pa, load_b_bf16(vtb, 32, 16, 0, lane), o1);

    // store ctx as [att][site][t][c] (c = h*32+d) so out-proj A is row-major
#pragma unroll
    for (int r = 0; r < 8; ++r) {
      int t = r + 8 * lh;
      int c0 = h * 32 + col;
      CTX[ctx_at(att, site, t, c0)]      = (__bf16)o0[r];
      CTX[ctx_at(att, site, t, c0 + 16)] = (__bf16)o1[r];
    }
  }
  __syncthreads();

  // ---------------- Phase 4: shared out-projection + bias + store -----------
  // exactly one (att, site) job per wave; A cached across all 16 N-tiles
  {
    int att  = wave >> 1;
    int site = wave & 1;
    const __bf16* Act = CTX + ctx_at(att, site, 0, 0);  // [16][256] row-major
    const __bf16* Wo  = wts + 6 * CC * CC;

    v8f acc[16];
#pragma unroll
    for (int i = 0; i < 16; ++i)
      acc[i] = v8f{0.f, 0.f, 0.f, 0.f, 0.f, 0.f, 0.f, 0.f};

    for (int ks = 0; ks < 8; ++ks) {
      int k0 = ks * 32;
      v16bf a = load_a_bf16(Act, CC, col, k0, lane);
#pragma unroll
      for (int nt = 0; nt < 16; ++nt) {
        v16bf b = load_b_bf16(Wo, CC, nt * 16, k0, lane);
        acc[nt] = wmma_bf16(a, b, acc[nt]);
      }
    }

    const long outbase  = (long)att * B_ * N_ * TT * CC;
    const long siteflat = (long)wg * SITES + site;
#pragma unroll
    for (int nt = 0; nt < 16; ++nt) {
      int c = nt * 16 + col;
      float bias = b_out[c];
#pragma unroll
      for (int r = 0; r < 8; ++r) {
        int t = r + 8 * lh;
        if (t < TT)
          out[outbase + (siteflat * TT + t) * CC + c] = acc[nt][r] + bias;
      }
    }
  }
}

// ---------- host launcher ----------
extern "C" void kernel_launch(void* const* d_in, const int* in_sizes, int n_in,
                              void* d_out, int out_size, void* d_ws, size_t ws_size,
                              hipStream_t stream) {
  (void)in_sizes; (void)n_in; (void)out_size; (void)ws_size;

  const float* x0 = (const float*)d_in[0];
  const float* x1 = (const float*)d_in[1];
  const float* x2 = (const float*)d_in[2];
  const float* x3 = (const float*)d_in[3];
  const float* x4 = (const float*)d_in[4];
  const float* x5 = (const float*)d_in[5];

  WPtrs wp;
  for (int i = 0; i < 7; ++i) wp.p[i] = (const float*)d_in[6 + i];
  __bf16* wbf = (__bf16*)d_ws;   // 7 * 65536 * 2B = 896 KB (lives in L2)

  const float* bo = (const float*)d_in[13];
  const float* kj = (const float*)d_in[14];
  const float* vf = (const float*)d_in[15];

  // Pre-pass: weights fp32 -> bf16 once per launch (deterministic).
  convert_weights_kernel<<<(7 * CC * CC) / NTHREADS, NTHREADS, 0, stream>>>(wp, wbf);

  // Fused attention: one WG per 2 (b,n) sites.
  int grid = (B_ * N_) / SITES;  // 2048
  flowspeed_fused_kernel<<<grid, NTHREADS, SMEM_BYTES, stream>>>(
      x0, x1, x2, x3, x4, x5, wbf, bo, kj, vf, (float*)d_out);
}
